// RecurrentEncoder_39616778338647
// MI455X (gfx1250) — matrix operand order repaired and checked by reading, not compiled, not run on hardware
//
#include <hip/hip_runtime.h>
#include <hip/hip_bf16.h>

// ---------------------------------------------------------------------------
// Types for CDNA5 WMMA
// ---------------------------------------------------------------------------
typedef __bf16 v16bf __attribute__((ext_vector_type(16)));
typedef float  v8f   __attribute__((ext_vector_type(8)));
typedef unsigned int u32x4 __attribute__((ext_vector_type(4)));

// global (address_space(1)) pointer types so weight fetches lower to
// global_load_b128 (LOADcnt only) instead of flat_load_b128 (LOAD+DS cnt)
typedef __attribute__((address_space(1))) const __bf16 gbf16;
typedef __attribute__((address_space(1))) const u32x4  gu32x4;

union AFrag { v16bf v; u32x4 q[2]; };
union BFrag { v16bf v; u32x4 q[2]; };

static constexpr int B_   = 256;
static constexpr int T_   = 128;
static constexpr int FIN_ = 16;
static constexpr int H_   = 512;
static constexpr int L_   = 3;
static constexpr int OUT_ = 16;
static constexpr int K2H  = 2 * H_;     // 1024 (concat input dim)
static constexpr int G4H  = 4 * H_;     // 2048 (gate dim)
static constexpr int PITCH = 520;       // bf16 elems per LDS row (1040B, 16B aligned, bank-skewed)

// float -> bf16 round-to-nearest-even, no reliance on cast codegen
static __device__ __forceinline__ __bf16 f2bf(float f) {
    union { float f; unsigned u; } in; in.f = f;
    unsigned u = in.u;
    unsigned r = u + 0x7FFFu + ((u >> 16) & 1u);
    union { unsigned short s; __bf16 b; } out;
    out.s = (unsigned short)(r >> 16);
    return out.b;
}

static __device__ __forceinline__ float sigf(float x) {
    return 1.0f / (1.0f + __expf(-x));
}
static __device__ __forceinline__ float tanh_fast(float x) {
    return 2.0f * sigf(2.0f * x) - 1.0f;
}

// ---------------------------------------------------------------------------
// Kernel 1: transpose + bf16-convert LSTM weights:  wT[l][n][k] = lstm_w[l][k][n]
// ---------------------------------------------------------------------------
__global__ __launch_bounds__(256) void k_transpose_w(const float* __restrict__ lstm_w,
                                                     __bf16* __restrict__ wT) {
    __shared__ float tile[16][17];
    int l  = blockIdx.z;
    int n0 = blockIdx.x * 16;
    int k0 = blockIdx.y * 16;
    int tx = threadIdx.x & 15;
    int ty = threadIdx.x >> 4;
    // coalesced read over n
    tile[ty][tx] = lstm_w[((size_t)l * K2H + (k0 + ty)) * G4H + (n0 + tx)];
    __syncthreads();
    // coalesced write over k
    wT[((size_t)l * G4H + (n0 + ty)) * K2H + (k0 + tx)] = f2bf(tile[tx][ty]);
}

// ---------------------------------------------------------------------------
// Kernel 2: input projection + relu, to bf16:  xp[t][b][h]
// ---------------------------------------------------------------------------
__global__ __launch_bounds__(256) void k_proj(const float* __restrict__ x,
                                              const float* __restrict__ proj_w,
                                              const float* __restrict__ proj_b,
                                              __bf16* __restrict__ xp) {
    int h = blockIdx.x * 256 + threadIdx.x;   // 0..511
    int t = blockIdx.y;
    int b = blockIdx.z;
    const float* xr = x + ((size_t)b * T_ + t) * FIN_;
    float s = proj_b[h];
#pragma unroll
    for (int f = 0; f < FIN_; ++f) s += xr[f] * proj_w[f * H_ + h];
    xp[((size_t)t * B_ + b) * H_ + h] = f2bf(fmaxf(s, 0.0f));
}

// ---------------------------------------------------------------------------
// Kernel 3: the 3-layer LSTM recurrence.
//   grid = 16 workgroups (one 16-row batch tile each), block = 512 (16 waves).
//   Wave w owns hidden columns [w*32, w*32+32): 8 WMMA accumulators (i,g,f,o x2).
//   h-state + c-state in LDS; accumulators + B fragments in registers.
// ---------------------------------------------------------------------------
__global__ __launch_bounds__(512, 1) void k_lstm(const __bf16* __restrict__ xp,
                                                 const __bf16* __restrict__ wT,
                                                 const float* __restrict__ lstm_b,
                                                 float* __restrict__ y) {
    __shared__ __align__(16) __bf16 s_in[16 * PITCH];       // current layer input (bf16)
    __shared__ __align__(16) __bf16 s_h[L_][16 * PITCH];    // recurrent h per layer (bf16)
    __shared__ float s_c[L_][16 * 32 * 16];                 // c state: [l][wave*512 + row*32 + col]

    const int tid  = threadIdx.x;
    const int w    = tid >> 5;          // wave 0..15
    const int lane = tid & 31;
    const int c16  = lane & 15;         // column-within-tile / row-within-tile selector
    const int hl   = lane >> 4;         // lane half
    const int b0   = blockIdx.x * 16;   // batch tile base

    // zero the recurrent h state and c state
    {
        unsigned short* p = (unsigned short*)&s_h[0][0];
        for (int i = tid; i < L_ * 16 * PITCH; i += 512) p[i] = 0;
        float* pc = &s_c[0][0];
        for (int i = tid; i < L_ * 16 * 32 * 16; i += 512) pc[i] = 0.0f;
    }

    // per-lane gate biases (depend on column only) -- scalars, added post-WMMA
    float bias[L_][2][4];
#pragma unroll
    for (int l = 0; l < L_; ++l)
#pragma unroll
        for (int j = 0; j < 2; ++j)
#pragma unroll
            for (int g = 0; g < 4; ++g)
                bias[l][j][g] = lstm_b[l * G4H + g * H_ + w * 32 + j * 16 + c16];

    __syncthreads();

    for (int t = 0; t < T_; ++t) {
        // stage xp[t] tile (16 rows x 512 bf16) into s_in
        {
            const __bf16* src = xp + ((size_t)t * B_ + b0) * H_;
            for (int i = tid; i < 1024; i += 512) {
                int r = i >> 6, c = i & 63;
                *(u32x4*)(s_in + r * PITCH + c * 8) =
                    *(const u32x4*)(src + (size_t)r * H_ + c * 8);
            }
        }
        __syncthreads();

#pragma unroll
        for (int l = 0; l < L_; ++l) {
            // Opacify the weight base pointer once per (t, layer): prevents LICM
            // from hoisting weight-fragment loads out of the time loop (which
            // previously spilled every B fragment to scratch). Cast the result
            // to an address_space(1) pointer so loads stay global_load_b128.
            unsigned long long wv =
                (unsigned long long)(wT + (size_t)l * G4H * K2H);
            asm volatile("" : "+s"(wv));
            gbf16* wl = (gbf16*)wv;

            // accumulators start at zero; bias is added in the epilogue
            v8f acc[2][4];
#pragma unroll
            for (int j = 0; j < 2; ++j)
#pragma unroll
                for (int g = 0; g < 4; ++g)
#pragma unroll
                    for (int e = 0; e < 8; ++e) acc[j][g][e] = 0.0f;

            // K loop: first half from s_in (layer input), second half from s_h[l]
#pragma unroll
            for (int srcsel = 0; srcsel < 2; ++srcsel) {
                const __bf16* abase = srcsel ? s_h[l] : s_in;
                const int kgbase = srcsel * H_;
                for (int kc = 0; kc < 16; ++kc) {
                    // A fragment from LDS
                    AFrag af;
                    const int aoff = c16 * PITCH + kc * 32 + hl * 8;
                    af.q[0] = *(const u32x4*)(abase + aoff);
                    af.q[1] = *(const u32x4*)(abase + aoff + 16);

                    // Batch-load ALL 8 B fragments first (distinct registers),
                    // so the waitcnt pass can overlap the load stream with the
                    // WMMA pipe using partial loadcnt waits.
                    const int kg = kgbase + kc * 32 + hl * 8;
                    BFrag bfr[2][4];
#pragma unroll
                    for (int j = 0; j < 2; ++j)
#pragma unroll
                        for (int g = 0; g < 4; ++g) {
                            const int n = w * 32 + j * 16 + g * H_ + c16;
                            gbf16* bp = wl + (size_t)n * K2H + kg;
                            bfr[j][g].q[0] = *(gu32x4*)(bp);
                            bfr[j][g].q[1] = *(gu32x4*)(bp + 16);
                        }
#pragma unroll
                    for (int j = 0; j < 2; ++j)
#pragma unroll
                        for (int g = 0; g < 4; ++g) {
                            acc[j][g] = __builtin_amdgcn_wmma_f32_16x16x32_bf16(
                                false, af.v, false, bfr[j][g].v, (short)0,
                                acc[j][g], false, false);
                        }
                    // Desired per-iteration schedule: A ds-reads, then the full
                    // B-fragment load batch, then the 8 WMMAs.
                    __builtin_amdgcn_sched_group_barrier(0x100, 2, 0);  // DS read
                    __builtin_amdgcn_sched_group_barrier(0x020, 16, 0); // VMEM read
                    __builtin_amdgcn_sched_group_barrier(0x008, 8, 0);  // WMMA
                }
            }

            __syncthreads();   // all waves done reading s_in / s_h[l]

            // elementwise LSTM cell update (C layout: row = e + 8*hl, col = tile col)
#pragma unroll
            for (int j = 0; j < 2; ++j) {
                const int colg = w * 32 + j * 16 + c16;
                float* cbase = &s_c[l][w * 512 + j * 16 + c16];
#pragma unroll
                for (int e = 0; e < 8; ++e) {
                    float iv = acc[j][0][e] + bias[l][j][0];
                    float gv = acc[j][1][e] + bias[l][j][1];
                    float fv = acc[j][2][e] + bias[l][j][2];
                    float ov = acc[j][3][e] + bias[l][j][3];
                    const int row = e + hl * 8;
                    float cp = cbase[row * 32];
                    float cn = sigf(fv + 1.0f) * cp + sigf(iv) * tanh_fast(gv);
                    float hn = sigf(ov) * tanh_fast(cn);
                    cbase[row * 32] = cn;
                    s_h[l][row * PITCH + colg] = f2bf(hn);
                    if (l < 2) {
                        s_in[row * PITCH + colg] = f2bf(fmaxf(hn, 0.0f));
                    } else {
                        y[((size_t)(b0 + row) * T_ + t) * H_ + colg] = hn;
                    }
                }
            }
            __syncthreads();   // writes visible before next matmul / staging
        }
    }
}

// ---------------------------------------------------------------------------
// Kernel 4: output projection  out[b][t][o] = y[b][t][:] @ out_w + out_b
// ---------------------------------------------------------------------------
__global__ __launch_bounds__(256) void k_out(const float* __restrict__ y,
                                             const float* __restrict__ out_w,
                                             const float* __restrict__ out_b,
                                             float* __restrict__ out) {
    __shared__ float s_w[H_ * OUT_];   // 32 KB
    for (int i = threadIdx.x; i < H_ * OUT_; i += 256) s_w[i] = out_w[i];
    __syncthreads();

    const int bt = blockIdx.x * 256 + threadIdx.x;   // 0..32767
    const float* yr = y + (size_t)bt * H_;
    float acc[OUT_];
#pragma unroll
    for (int o = 0; o < OUT_; ++o) acc[o] = out_b[o];
    for (int h = 0; h < H_; ++h) {
        float yv = yr[h];
        const float4* wr = (const float4*)(s_w + h * OUT_);
#pragma unroll
        for (int q = 0; q < 4; ++q) {
            float4 wv = wr[q];
            acc[q * 4 + 0] += yv * wv.x;
            acc[q * 4 + 1] += yv * wv.y;
            acc[q * 4 + 2] += yv * wv.z;
            acc[q * 4 + 3] += yv * wv.w;
        }
    }
    float4* op = (float4*)(out + (size_t)bt * OUT_);
#pragma unroll
    for (int q = 0; q < 4; ++q)
        op[q] = make_float4(acc[q * 4 + 0], acc[q * 4 + 1], acc[q * 4 + 2], acc[q * 4 + 3]);
}

// ---------------------------------------------------------------------------
// Launch
// ---------------------------------------------------------------------------
extern "C" void kernel_launch(void* const* d_in, const int* in_sizes, int n_in,
                              void* d_out, int out_size, void* d_ws, size_t ws_size,
                              hipStream_t stream) {
    (void)in_sizes; (void)n_in; (void)out_size; (void)ws_size;

    const float* x      = (const float*)d_in[0];
    const float* proj_w = (const float*)d_in[1];
    const float* proj_b = (const float*)d_in[2];
    const float* lstm_w = (const float*)d_in[3];
    const float* lstm_b = (const float*)d_in[4];
    const float* out_w  = (const float*)d_in[5];
    const float* out_b  = (const float*)d_in[6];
    float* out = (float*)d_out;

    // workspace partition
    char* ws = (char*)d_ws;
    __bf16* wT = (__bf16*)ws;                                   // 3*2048*1024*2  = 12,582,912 B
    size_t off = (size_t)L_ * G4H * K2H * sizeof(__bf16);
    __bf16* xp = (__bf16*)(ws + off);                           // 128*256*512*2  = 33,554,432 B
    off += (size_t)T_ * B_ * H_ * sizeof(__bf16);
    float* yb = (float*)(ws + off);                             // 256*128*512*4  = 67,108,864 B

    // 1) weight transpose + bf16 convert
    {
        dim3 grid(G4H / 16, K2H / 16, L_);
        k_transpose_w<<<grid, 256, 0, stream>>>(lstm_w, wT);
    }
    // 2) input projection
    {
        dim3 grid(H_ / 256, T_, B_);
        k_proj<<<grid, 256, 0, stream>>>(x, proj_w, proj_b, xp);
    }
    // 3) LSTM recurrence (WMMA core)
    {
        k_lstm<<<B_ / 16, 512, 0, stream>>>(xp, wT, lstm_b, yb);
    }
    // 4) output projection
    {
        k_out<<<(B_ * T_) / 256, 256, 0, stream>>>(yb, out_w, out_b, out);
    }
}
